// CompetitiveLayer_89644557402931
// MI455X (gfx1250) — matrix operand order repaired and checked by reading, not compiled, not run on hardware
//
#include <hip/hip_runtime.h>
#include <hip/hip_bf16.h>
#include <math.h>

typedef __attribute__((ext_vector_type(16))) __bf16 v16bf;
typedef __attribute__((ext_vector_type(8)))  __bf16 v8bf;
typedef __attribute__((ext_vector_type(8)))  float  v8f;

#define TILE_M 128
#define TILE_N 128
#define TILE_K 32
#define LDK    40      // 32 + 8 pad: lane stride 80B = 20 dwords -> conflict-free b128 LDS loads
#define DIM    2048
#define KSEL   16
#define INV_T  5.0f    // 1 / 0.2

// ---------------------------------------------------------------------------
// Kernel 1: per-row reciprocal L2 norm of x
// ---------------------------------------------------------------------------
__global__ __launch_bounds__(256) void rownorm_kernel(const float* __restrict__ x,
                                                      float* __restrict__ rscale,
                                                      int D) {
    const int row = blockIdx.x;
    const int t   = threadIdx.x;
    const float* xr = x + (size_t)row * D;
    float s = 0.0f;
    for (int c = t; c < D; c += 256) {
        float v = xr[c];
        s += v * v;
    }
    __shared__ float red[256];
    red[t] = s;
    __syncthreads();
    for (int off = 128; off > 0; off >>= 1) {
        if (t < off) red[t] += red[t + off];
        __syncthreads();
    }
    if (t == 0) rscale[row] = 1.0f / fmaxf(sqrtf(red[0]), 1e-12f);
}

// ---------------------------------------------------------------------------
// Kernel 2: sims = x @ protos^T via v_wmma_f32_16x16x32_bf16
// ---------------------------------------------------------------------------
__device__ __forceinline__ void cvt_store16(const float* __restrict__ g, __bf16* s) {
    const float4* g4 = (const float4*)g;
    float4 v0 = g4[0], v1 = g4[1], v2 = g4[2], v3 = g4[3];
    v8bf o0, o1;
    o0[0] = (__bf16)v0.x; o0[1] = (__bf16)v0.y; o0[2] = (__bf16)v0.z; o0[3] = (__bf16)v0.w;
    o0[4] = (__bf16)v1.x; o0[5] = (__bf16)v1.y; o0[6] = (__bf16)v1.z; o0[7] = (__bf16)v1.w;
    o1[0] = (__bf16)v2.x; o1[1] = (__bf16)v2.y; o1[2] = (__bf16)v2.z; o1[3] = (__bf16)v2.w;
    o1[4] = (__bf16)v3.x; o1[5] = (__bf16)v3.y; o1[6] = (__bf16)v3.z; o1[7] = (__bf16)v3.w;
    *(v8bf*)s       = o0;
    *(v8bf*)(s + 8) = o1;
}

__global__ __launch_bounds__(256) void gemm_wmma_kernel(const float* __restrict__ X,
                                                        const float* __restrict__ P,
                                                        float* __restrict__ S,
                                                        int Pn, int D) {
    __shared__ alignas(16) __bf16 As[2][TILE_M][LDK];
    __shared__ alignas(16) __bf16 Bs[2][TILE_N][LDK];

    const int t    = threadIdx.x;
    const int lane = t & 31;
    const int wave = t >> 5;        // 0..7
    const int wm   = wave & 3;      // m offset = 32*wm
    const int wn   = wave >> 2;     // n offset = 64*wn
    const int l    = lane & 15;
    const int h    = lane >> 4;

    const int mBase = blockIdx.y * TILE_M;
    const int nBase = blockIdx.x * TILE_N;

    // Cooperative staging: thread t loads 16 contiguous fp32 of one row segment
    const int ldRow = t >> 1;          // 0..127
    const int ldKs  = (t & 1) * 16;    // 0 or 16

    const int KITER = D / TILE_K;      // 64

    v8f acc[2][4];
#pragma unroll
    for (int mi = 0; mi < 2; ++mi)
#pragma unroll
        for (int ni = 0; ni < 4; ++ni)
            acc[mi][ni] = (v8f){};

    // Prologue stage kk = 0 into buffer 0
    cvt_store16(X + (size_t)(mBase + ldRow) * D + ldKs, &As[0][ldRow][ldKs]);
    cvt_store16(P + (size_t)(nBase + ldRow) * D + ldKs, &Bs[0][ldRow][ldKs]);
    __syncthreads();

    for (int kk = 0; kk < KITER; ++kk) {
        const int buf = kk & 1;
        if (kk + 1 < KITER) {
            const int koff = (kk + 1) * TILE_K;
            cvt_store16(X + (size_t)(mBase + ldRow) * D + koff + ldKs, &As[buf ^ 1][ldRow][ldKs]);
            cvt_store16(P + (size_t)(nBase + ldRow) * D + koff + ldKs, &Bs[buf ^ 1][ldRow][ldKs]);
        }

        // A fragments (16x32 bf16, ISA layout: lane half h holds K {8h..8h+7} and {16+8h..16+8h+7})
        v16bf afr[2];
#pragma unroll
        for (int mi = 0; mi < 2; ++mi) {
            const __bf16* ap = &As[buf][wm * 32 + mi * 16 + l][0];
            v8bf lo = *(const v8bf*)(ap + h * 8);
            v8bf hi = *(const v8bf*)(ap + 16 + h * 8);
            afr[mi] = __builtin_shufflevector(lo, hi, 0,1,2,3,4,5,6,7,8,9,10,11,12,13,14,15);
        }
        // B fragments (32x16 bf16: lane = column N, half h holds K {8h..8h+7} and {16+8h..16+8h+7})
        v16bf bfr[4];
#pragma unroll
        for (int ni = 0; ni < 4; ++ni) {
            const __bf16* bp = &Bs[buf][wn * 64 + ni * 16 + l][0] + h * 8;
            v8bf lo = *(const v8bf*)(bp);
            v8bf hi = *(const v8bf*)(bp + 16);
            bfr[ni] = __builtin_shufflevector(lo, hi, 0,1,2,3,4,5,6,7,8,9,10,11,12,13,14,15);
        }

#pragma unroll
        for (int mi = 0; mi < 2; ++mi)
#pragma unroll
            for (int ni = 0; ni < 4; ++ni)
                acc[mi][ni] = __builtin_amdgcn_wmma_f32_16x16x32_bf16(
                    false, afr[mi], false, bfr[ni], (short)0, acc[mi][ni], false, false);

        __syncthreads();
    }

    // Epilogue: C/D layout -> element r of lane (l,h) is row (8h + r), col l
#pragma unroll
    for (int mi = 0; mi < 2; ++mi) {
#pragma unroll
        for (int ni = 0; ni < 4; ++ni) {
            const int gm = mBase + wm * 32 + mi * 16 + h * 8;
            const int gn = nBase + wn * 64 + ni * 16 + l;
#pragma unroll
            for (int r = 0; r < 8; ++r)
                S[(size_t)(gm + r) * Pn + gn] = acc[mi][ni][r];
        }
    }
}

// ---------------------------------------------------------------------------
// Kernel 3: per-row top-16 -> softmax(v/T) -> sparse scatter (in place on sims)
// ---------------------------------------------------------------------------
__global__ __launch_bounds__(256) void topk_kernel(float* __restrict__ sims,
                                                   const float* __restrict__ rscale,
                                                   int Pn) {
    const int row = blockIdx.x;
    const int t   = threadIdx.x;

    __shared__ float rowbuf[4096];
    __shared__ float redv[256];
    __shared__ int   redi[256];
    __shared__ float topv[KSEL];
    __shared__ int   topi[KSEL];
    __shared__ float probs[KSEL];

    float* srow = sims + (size_t)row * Pn;
    const float s = rscale[row];

#pragma unroll
    for (int i = 0; i < 16; ++i) {
        int col = t + 256 * i;
        rowbuf[col] = srow[col] * s;
    }
    __syncthreads();

    for (int sel = 0; sel < KSEL; ++sel) {
        float bv = -INFINITY;
        int   bi = 0x7fffffff;
#pragma unroll
        for (int i = 0; i < 16; ++i) {
            int col = t + 256 * i;
            float v = rowbuf[col];
            if (v > bv) { bv = v; bi = col; }   // ascending col -> lowest index on ties
        }
        redv[t] = bv;
        redi[t] = bi;
        __syncthreads();
        for (int off = 128; off > 0; off >>= 1) {
            if (t < off) {
                float v2 = redv[t + off];
                int   i2 = redi[t + off];
                if (v2 > redv[t] || (v2 == redv[t] && i2 < redi[t])) {
                    redv[t] = v2;
                    redi[t] = i2;
                }
            }
            __syncthreads();
        }
        if (t == 0) {
            topv[sel] = redv[0];
            topi[sel] = redi[0];
            rowbuf[redi[0]] = -INFINITY;
        }
        __syncthreads();
    }

    if (t == 0) {
        float m = topv[0];          // selections are descending -> topv[0] is max
        float e[KSEL];
        float sum = 0.0f;
#pragma unroll
        for (int j = 0; j < KSEL; ++j) {
            e[j] = __expf((topv[j] - m) * INV_T);
            sum += e[j];
        }
        float inv = 1.0f / sum;
#pragma unroll
        for (int j = 0; j < KSEL; ++j) probs[j] = e[j] * inv;
    }
    __syncthreads();

    // Full-row write: zeros everywhere except the selected 16 columns
#pragma unroll
    for (int i = 0; i < 16; ++i) {
        int col = t + 256 * i;
        float v = 0.0f;
#pragma unroll
        for (int j = 0; j < KSEL; ++j)
            if (topi[j] == col) v = probs[j];
        srow[col] = v;
    }
}

// ---------------------------------------------------------------------------
// Launch
// ---------------------------------------------------------------------------
extern "C" void kernel_launch(void* const* d_in, const int* in_sizes, int n_in,
                              void* d_out, int out_size, void* d_ws, size_t ws_size,
                              hipStream_t stream) {
    const float* x      = (const float*)d_in[0];
    const float* protos = (const float*)d_in[1];
    const int D  = DIM;
    const int Bn = in_sizes[0] / D;   // 16384
    const int Pn = in_sizes[1] / D;   // 4096

    float* sims   = (float*)d_out;    // d_out doubles as the sims buffer
    float* rscale = (float*)d_ws;     // 64 KB of workspace

    rownorm_kernel<<<Bn, 256, 0, stream>>>(x, rscale, D);

    dim3 grid(Pn / TILE_N, Bn / TILE_M);
    gemm_wmma_kernel<<<grid, 256, 0, stream>>>(x, protos, sims, Pn, D);

    topk_kernel<<<Bn, 256, 0, stream>>>(sims, rscale, Pn);
}
